// ObjectConceptMoELayer_53412213293899
// MI455X (gfx1250) — compile-verified, matrix operand order checked by reading, not run on hardware
//
#include <hip/hip_runtime.h>
#include <hip/hip_bf16.h>

// Problem dims (fixed by setup_inputs): B=4, S=2048 -> T=8192; H=1024; F=2048;
// Ei=32 sparse experts (top-16 of 32), Es=4 shared experts -> E=36 combined.
#define kT 8192
#define kH 1024
#define kF 2048
#define kEi 32
#define kEs 4
#define kE 36
#define kTopK 16
#define kBM 64    // tokens per block in the MoE GEMM kernel
#define kFC 1024  // F-chunk width (kF / 2): h computed and consumed per chunk

typedef __attribute__((ext_vector_type(16))) __bf16 v16bf;
typedef __attribute__((ext_vector_type(8)))  float  v8f;

union Frag {
  v16bf    v;
  unsigned u[8];
};

// round-to-nearest-even fp32 -> bf16
__device__ __forceinline__ unsigned short f2bf(float f) {
  unsigned u = __float_as_uint(f);
  u += 0x7FFFu + ((u >> 16) & 1u);
  return (unsigned short)(u >> 16);
}
__device__ __forceinline__ unsigned pack2(float lo, float hi) {
  return (unsigned)f2bf(lo) | ((unsigned)f2bf(hi) << 16);
}

// 16-bit 16x32 A/B fragment (ISA 7.12.2): lane half h owns K = h*8 + {0..7, 16..23};
// with K-contiguous bf16 storage each half is two 16-byte runs -> 2x b128 loads.
// `S` is a uint4* view of a bf16 matrix with row length `rowlen` elements;
// fragment rows are `row + lane%16`.
__device__ __forceinline__ Frag load_frag16(const uint4* __restrict__ S, int rowlen,
                                            int row, int kbase, int lane) {
  const int rr   = row + (lane & 15);
  const int half = (lane >> 4) & 1;
  const int base = rr * rowlen + kbase + half * 8;  // bf16 element index, %8 == 0
  const uint4 lo = S[(base >> 3) + 0];              // K = base .. base+7
  const uint4 hi = S[(base >> 3) + 2];              // K = base+16 .. base+23
  Frag f;
  f.u[0] = lo.x; f.u[1] = lo.y; f.u[2] = lo.z; f.u[3] = lo.w;
  f.u[4] = hi.x; f.u[5] = hi.y; f.u[6] = hi.z; f.u[7] = hi.w;
  return f;
}

// ---------------------------------------------------------------------------
// Weight prep: transpose [K][N] fp32 -> [N][K] bf16 (K contiguous), per matrix.
// 32x32 LDS tile; coalesced reads and writes. One-pass streaming traffic.
__global__ __launch_bounds__(256)
void transpose_bf16_kernel(const float* __restrict__ src, unsigned short* __restrict__ dst,
                           int K, int N) {
  __shared__ float tile[32][33];
  const int e = blockIdx.z;
  src += (size_t)e * K * N;
  dst += (size_t)e * K * N;
  const int n0 = blockIdx.x * 32;
  const int k0 = blockIdx.y * 32;
  const int tx = threadIdx.x;   // 0..31
  const int ty = threadIdx.y;   // 0..7
#pragma unroll
  for (int r = ty; r < 32; r += 8)
    tile[r][tx] = src[(size_t)(k0 + r) * N + n0 + tx];
  __syncthreads();
#pragma unroll
  for (int r = ty; r < 32; r += 8)
    dst[(size_t)(n0 + r) * K + k0 + tx] = f2bf(tile[tx][r]);
}

// ---------------------------------------------------------------------------
// Zero output + aux accumulators (harness poisons buffers with 0xAA).
__global__ void zero_kernel(float* __restrict__ out, size_t n, float* __restrict__ acc, int na) {
  size_t i = (size_t)blockIdx.x * blockDim.x + threadIdx.x;
  const size_t stride = (size_t)gridDim.x * blockDim.x;
  for (; i < n; i += stride) out[i] = 0.0f;
  if (blockIdx.x == 0 && threadIdx.x < (unsigned)na) acc[threadIdx.x] = 0.0f;
}

// ---------------------------------------------------------------------------
// Gating: logits, dense softmax (for aux), top-16 renormalized weights,
// shared-expert softmax. Writes comb[T][36]; accumulates per-expert sums.
__global__ __launch_bounds__(256)
void gate_kernel(const float* __restrict__ x, const float* __restrict__ gi,
                 const float* __restrict__ gs, float* __restrict__ comb,
                 float* __restrict__ sumInd, float* __restrict__ sumProb) {
  const int t = blockIdx.x * blockDim.x + threadIdx.x;
  if (t >= kT) return;
  float li[kEi], ls[kEs];
#pragma unroll
  for (int e = 0; e < kEi; ++e) li[e] = 0.0f;
#pragma unroll
  for (int e = 0; e < kEs; ++e) ls[e] = 0.0f;
  const float* xr = x + (size_t)t * kH;
  for (int h = 0; h < kH; ++h) {
    const float xv = xr[h];
    const float* g = gi + (size_t)h * kEi;
#pragma unroll
    for (int e = 0; e < kEi; ++e) li[e] += xv * g[e];
    const float* g2 = gs + (size_t)h * kEs;
#pragma unroll
    for (int e = 0; e < kEs; ++e) ls[e] += xv * g2[e];
  }
  // dense softmax over the 32 sparse-expert logits (for aux loss)
  float pm = li[0];
#pragma unroll
  for (int e = 1; e < kEi; ++e) pm = fmaxf(pm, li[e]);
  float pr[kEi];
  float ps = 0.0f;
#pragma unroll
  for (int e = 0; e < kEi; ++e) { pr[e] = __expf(li[e] - pm); ps += pr[e]; }
  // top-16 of 32 (the max logit is always selected, so max-over-selected == pm)
  unsigned sel = 0u;
  for (int it = 0; it < kTopK; ++it) {
    int bi = 0;
    float bv = -3.4e38f;
    for (int e = 0; e < kEi; ++e) {
      const bool taken = (sel >> e) & 1u;
      if (!taken && li[e] > bv) { bv = li[e]; bi = e; }
    }
    sel |= 1u << bi;
  }
  float ssum = 0.0f;
#pragma unroll
  for (int e = 0; e < kEi; ++e)
    if ((sel >> e) & 1u) ssum += pr[e];
  const float inv_ssum = 1.0f / ssum, inv_ps = 1.0f / ps;
  float* cr = comb + (size_t)t * kE;
#pragma unroll
  for (int e = 0; e < kEi; ++e) {
    const bool on = (sel >> e) & 1u;
    cr[e] = on ? pr[e] * inv_ssum : 0.0f;
    atomicAdd(&sumInd[e], on ? 1.0f : 0.0f);
    atomicAdd(&sumProb[e], pr[e] * inv_ps);
  }
  // shared (dense) experts: softmax over 4 logits
  float sm = ls[0];
#pragma unroll
  for (int e = 1; e < kEs; ++e) sm = fmaxf(sm, ls[e]);
  float es[kEs], ss = 0.0f;
#pragma unroll
  for (int e = 0; e < kEs; ++e) { es[e] = __expf(ls[e] - sm); ss += es[e]; }
#pragma unroll
  for (int e = 0; e < kEs; ++e) cr[kEi + e] = es[e] / ss;
}

// ---------------------------------------------------------------------------
// Fused per-(token-tile, expert) MoE block, 64 tokens x 256 threads (8 waves).
// F is processed in two 1024-wide chunks to keep h in LDS while maximizing M
// (more token reuse per weight byte -> half the L2 weight traffic vs M=32):
//   for fc in {0,1}:
//     h_c = relu(x_tile @ W1_e[:, fc] + b1[fc])     [64 x 1024 bf16 in LDS]
//     y  += comb[t,e] * (h_c @ W2_e[fc, :] + (fc==0 ? b2 : 0))   (global f32 atomics)
// Weights come pre-transposed bf16 [e][n][k]; hot loop is pure b128 loads +
// v_wmma_f32_16x16x32_bf16.
__global__ __launch_bounds__(256, 1)
void moe_kernel(const float* __restrict__ x,
                const unsigned short* __restrict__ W1t,  // [36][kF][kH] bf16 (n-major)
                const unsigned short* __restrict__ W2t,  // [36][kH][kF] bf16 (n-major)
                const float* __restrict__ b1_ind, const float* __restrict__ b2_ind,
                const float* __restrict__ b1_sh,  const float* __restrict__ b2_sh,
                const float* __restrict__ comb,
                float* __restrict__ out) {
  extern __shared__ unsigned short smem[];
  unsigned short* xs = smem;              // [kBM][kH]  bf16 (128 KB), K-major
  unsigned short* hs = smem + kBM * kH;   // [kBM][kFC] bf16 (128 KB), K-major
  const uint4* xs4 = (const uint4*)xs;
  const uint4* hs4 = (const uint4*)hs;

  const int tb = blockIdx.x;
  const int e  = blockIdx.y;
  const int t0 = tb * kBM;
  const int tid  = threadIdx.x;
  const int lane = tid & 31;
  const int wave = tid >> 5;              // 0..7
  const int half = (lane >> 4) & 1;

  const uint4* W1e = (const uint4*)(W1t + (size_t)e * kF * kH);
  const uint4* W2e = (const uint4*)(W2t + (size_t)e * kF * kH);
  const float* B1 = (e < kEi) ? (b1_ind + (size_t)e * kF) : (b1_sh + (size_t)(e - kEi) * kF);
  const float* B2 = (e < kEi) ? (b2_ind + (size_t)e * kH) : (b2_sh + (size_t)(e - kEi) * kH);

  // Per-lane combine weights for the 4 M-tiles this block owns.
  float wtok[4][8];
#pragma unroll
  for (int mt = 0; mt < 4; ++mt)
#pragma unroll
    for (int r = 0; r < 8; ++r)
      wtok[mt][r] = comb[(size_t)(t0 + mt * 16 + r + 8 * half) * kE + e];

  // Stage x tile -> bf16 LDS (float4 global reads, packed uint2 LDS writes).
  for (int i = tid; i < (kBM * kH) / 4; i += 256) {
    const int idx = i * 4;
    const int m = idx >> 10;      // / kH
    const int k = idx & (kH - 1);
    const float4 xv = *(const float4*)(x + (size_t)(t0 + m) * kH + k);
    ((uint2*)xs)[i] = make_uint2(pack2(xv.x, xv.y), pack2(xv.z, xv.w));
  }
  __syncthreads();

  for (int fc = 0; fc < kF / kFC; ++fc) {
    // ---- GEMM 1: hs[kBM, kFC] = relu(x @ W1[:, fc*kFC ..] + b1) ----
    // 64 N-tiles in this chunk; each wave owns 8, processed 2 at a time; 4 M-tiles.
    for (int tg = wave * 8; tg < wave * 8 + 8; tg += 2) {
      v8f acc[4][2];
#pragma unroll
      for (int mt = 0; mt < 4; ++mt)
#pragma unroll
        for (int j = 0; j < 2; ++j)
#pragma unroll
          for (int i = 0; i < 8; ++i) acc[mt][j][i] = 0.0f;
      for (int kb = 0; kb < kH; kb += 32) {
        Frag a[4];
#pragma unroll
        for (int mt = 0; mt < 4; ++mt)
          a[mt] = load_frag16(xs4, kH, mt * 16, kb, lane);
#pragma unroll
        for (int j = 0; j < 2; ++j) {
          const Frag b = load_frag16(W1e, kH, fc * kFC + (tg + j) * 16, kb, lane);
#pragma unroll
          for (int mt = 0; mt < 4; ++mt)
            acc[mt][j] = __builtin_amdgcn_wmma_f32_16x16x32_bf16(
                false, a[mt].v, false, b.v, (short)0, acc[mt][j], false, false);
        }
      }
#pragma unroll
      for (int j = 0; j < 2; ++j) {
        const int nl = (tg + j) * 16 + (lane & 15);   // column within chunk
        const float bias = B1[fc * kFC + nl];
#pragma unroll
        for (int mt = 0; mt < 4; ++mt)
#pragma unroll
          for (int r = 0; r < 8; ++r) {
            const int m = mt * 16 + r + 8 * half;     // C/D layout: M = r + 8*(lane>=16)
            float vv = acc[mt][j][r] + bias;
            vv = vv > 0.0f ? vv : 0.0f;
            hs[m * kFC + nl] = f2bf(vv);
          }
      }
    }
    __syncthreads();

    // ---- GEMM 2: y_tile += w[t,e] * (hs @ W2[fc, :] + (fc==0 ? b2 : 0)) ----
    // 64 N-tiles (H/16); each wave owns 8, processed 2 at a time; 4 M-tiles.
    for (int tg = wave * 8; tg < wave * 8 + 8; tg += 2) {
      v8f acc[4][2];
#pragma unroll
      for (int mt = 0; mt < 4; ++mt)
#pragma unroll
        for (int j = 0; j < 2; ++j)
#pragma unroll
          for (int i = 0; i < 8; ++i) acc[mt][j][i] = 0.0f;
      for (int kb = 0; kb < kFC; kb += 32) {
        Frag a[4];
#pragma unroll
        for (int mt = 0; mt < 4; ++mt)
          a[mt] = load_frag16(hs4, kFC, mt * 16, kb, lane);
#pragma unroll
        for (int j = 0; j < 2; ++j) {
          const Frag b = load_frag16(W2e, kF, (tg + j) * 16, fc * kFC + kb, lane);
#pragma unroll
          for (int mt = 0; mt < 4; ++mt)
            acc[mt][j] = __builtin_amdgcn_wmma_f32_16x16x32_bf16(
                false, a[mt].v, false, b.v, (short)0, acc[mt][j], false, false);
        }
      }
#pragma unroll
      for (int j = 0; j < 2; ++j) {
        const int n = (tg + j) * 16 + (lane & 15);
        const float bias = (fc == 0) ? B2[n] : 0.0f;
#pragma unroll
        for (int mt = 0; mt < 4; ++mt)
#pragma unroll
          for (int r = 0; r < 8; ++r) {
            const int m = mt * 16 + r + 8 * half;
            atomicAdd(out + (size_t)(t0 + m) * kH + n, wtok[mt][r] * (acc[mt][j][r] + bias));
          }
      }
    }
    __syncthreads();
  }
}

// ---------------------------------------------------------------------------
// aux = E * sum_e (mean_t 1[selected]) * (mean_t softmax prob); shared part is 0.
__global__ void aux_kernel(const float* __restrict__ sumInd,
                           const float* __restrict__ sumProb,
                           float* __restrict__ out_aux, float invT) {
  if (threadIdx.x == 0) {
    float s = 0.0f;
    for (int e = 0; e < kEi; ++e) s += (sumInd[e] * invT) * (sumProb[e] * invT);
    *out_aux = (float)kEi * s;
  }
}

extern "C" void kernel_launch(void* const* d_in, const int* in_sizes, int n_in,
                              void* d_out, int out_size, void* d_ws, size_t ws_size,
                              hipStream_t stream) {
  (void)in_sizes; (void)n_in; (void)out_size; (void)ws_size;
  const float* x        = (const float*)d_in[0];
  const float* gate_ind = (const float*)d_in[1];
  const float* W1_ind   = (const float*)d_in[2];
  const float* b1_ind   = (const float*)d_in[3];
  const float* W2_ind   = (const float*)d_in[4];
  const float* b2_ind   = (const float*)d_in[5];
  const float* gate_sh  = (const float*)d_in[6];
  const float* W1_sh    = (const float*)d_in[7];
  const float* b1_sh    = (const float*)d_in[8];
  const float* W2_sh    = (const float*)d_in[9];
  const float* b2_sh    = (const float*)d_in[10];
  float* out = (float*)d_out;

  // Workspace: W1t bf16 [36][kF][kH] | W2t bf16 [36][kH][kF] | comb[T][36] f32 | 64 sums.
  const size_t wsz = (size_t)kE * kH * kF;          // elements per weight bank
  unsigned short* W1t = (unsigned short*)d_ws;
  unsigned short* W2t = W1t + wsz;
  float* comb    = (float*)(W2t + wsz);
  float* sumInd  = comb + (size_t)kT * kE;
  float* sumProb = sumInd + kEi;

  // 1) zero output + aux accumulators
  const size_t out_elems = (size_t)kT * kH + 1;     // y + aux scalar
  zero_kernel<<<512, 256, 0, stream>>>(out, out_elems, sumInd, kEi + kEi);

  // 2) weight prep: fp32 [K][N] -> bf16 [N][K]
  dim3 tb(32, 8);
  transpose_bf16_kernel<<<dim3(kF / 32, kH / 32, kEi), tb, 0, stream>>>(W1_ind, W1t, kH, kF);
  transpose_bf16_kernel<<<dim3(kF / 32, kH / 32, kEs), tb, 0, stream>>>(
      W1_sh, W1t + (size_t)kEi * kH * kF, kH, kF);
  transpose_bf16_kernel<<<dim3(kH / 32, kF / 32, kEi), tb, 0, stream>>>(W2_ind, W2t, kF, kH);
  transpose_bf16_kernel<<<dim3(kH / 32, kF / 32, kEs), tb, 0, stream>>>(
      W2_sh, W2t + (size_t)kEi * kH * kF, kF, kH);

  // 3) gating
  gate_kernel<<<kT / 256, 256, 0, stream>>>(x, gate_ind, gate_sh, comb, sumInd, sumProb);

  // 4) fused MoE GEMMs
  const size_t lds_bytes = (size_t)(kBM * kH + kBM * kFC) * sizeof(unsigned short);  // 256 KB
  hipFuncSetAttribute((const void*)moe_kernel,
                      hipFuncAttributeMaxDynamicSharedMemorySize, (int)lds_bytes);
  dim3 grid(kT / kBM, kE);
  moe_kernel<<<grid, 256, lds_bytes, stream>>>(x, W1t, W2t, b1_ind, b2_ind, b1_sh, b2_sh,
                                               comb, out);

  // 5) aux loss scalar
  aux_kernel<<<1, 32, 0, stream>>>(sumInd, sumProb, out + (size_t)kT * kH, 1.0f / (float)kT);
}
